// Attention_47605417509033
// MI455X (gfx1250) — matrix operand order repaired and checked by reading, not compiled, not run on hardware
//
#include <hip/hip_runtime.h>
#include <hip/hip_bf16.h>

// ---------------------------------------------------------------------------
// Attention layer for MI455X (gfx1250, wave32, WMMA).
//   B=2, S=2048, HIDDEN=2048, NUM_HEADS=16, HEAD_DIM=128.
// Pipeline: fp32->bf16 convert, 3x bf16 WMMA GEMM (QKV proj), RoPE,
//           flash-attention (WMMA for QK^T and PV), bf16 WMMA GEMM (Wo, f32 out)
// GEMM uses double-buffered LDS with GLOBAL_LOAD_ASYNC_TO_LDS_B128 for the
// A tile (ASYNCcnt path) overlapped with WMMA compute.
// ---------------------------------------------------------------------------

typedef unsigned short u16;
typedef __attribute__((ext_vector_type(16))) __bf16 v16bf;
typedef __attribute__((ext_vector_type(8)))  float  v8f;

#define NTOK   4096   // B*S
#define HID    2048
#define SEQ    2048
#define NHEAD  16
#define HDIM   128

static __device__ __forceinline__ u16 f2bf(float f) {
    unsigned int u = __float_as_uint(f);
    unsigned int r = u + 0x7FFFu + ((u >> 16) & 1u);   // round-to-nearest-even
    return (u16)(r >> 16);
}
static __device__ __forceinline__ float bf2f(u16 h) {
    return __uint_as_float(((unsigned int)h) << 16);
}

static __device__ __forceinline__ v16bf frag_from2(uint4 lo, uint4 hi) {
    union { uint4 u[2]; v16bf v; } t;
    t.u[0] = lo; t.u[1] = hi;
    return t.v;
}

static __device__ __forceinline__ v8f wmma_bf16(v16bf a, v16bf b, v8f c) {
    // D(16x16,f32) = A(16x32,bf16) x B(32x16,bf16) + C
    return __builtin_amdgcn_wmma_f32_16x16x32_bf16(
        false, a, false, b, (short)0, c, false, false);
}

// Async copy 16 bytes global -> LDS (ASYNCcnt-tracked, bypasses VGPRs).
static __device__ __forceinline__ void async_g2l_b128(unsigned lds_off,
                                                      const void* gaddr) {
    asm volatile("global_load_async_to_lds_b128 %0, %1, off"
                 :: "v"(lds_off), "v"(gaddr) : "memory");
}
static __device__ __forceinline__ void wait_asynccnt0() {
    asm volatile("s_wait_asynccnt 0x0" ::: "memory");
}

// ---------------------------------------------------------------------------
// Kernel 1: fp32 -> bf16, 8 elements per thread (n must be multiple of 8)
// ---------------------------------------------------------------------------
__global__ __launch_bounds__(256) void cvt_f32_bf16(const float* __restrict__ in,
                                                    u16* __restrict__ out, int n8) {
    int i = blockIdx.x * blockDim.x + threadIdx.x;
    if (i >= n8) return;
    const float4* p = reinterpret_cast<const float4*>(in) + (size_t)i * 2;
    float4 f0 = p[0], f1 = p[1];
    union { u16 e[8]; uint4 v; } o;
    o.e[0] = f2bf(f0.x); o.e[1] = f2bf(f0.y); o.e[2] = f2bf(f0.z); o.e[3] = f2bf(f0.w);
    o.e[4] = f2bf(f1.x); o.e[5] = f2bf(f1.y); o.e[6] = f2bf(f1.z); o.e[7] = f2bf(f1.w);
    reinterpret_cast<uint4*>(out)[i] = o.v;
}

// ---------------------------------------------------------------------------
// Kernel 2: GEMM  C[4096,2048] = A[4096,2048] * W[2048,2048]  (bf16 in, f32
// accum).  Block tile 128x128, 256 threads = 8 waves, each wave 32x64
// (2x4 WMMA accumulators).  K staged 32 at a time, double-buffered LDS:
// A tile via async-to-LDS, B tile via registers (transpose scatter), both
// for tile kt+1 issued before the WMMA work of tile kt.
// ---------------------------------------------------------------------------
__global__ __launch_bounds__(256) void gemm_bf16(const u16* __restrict__ A,
                                                 const u16* __restrict__ W,
                                                 u16* __restrict__ Cb,
                                                 float* __restrict__ Cf,
                                                 int out_f32) {
    __shared__ alignas(16) u16 As[2][128][40];   // [m][k], 80B row stride
    __shared__ alignas(16) u16 Bs[2][128][40];   // [n][k] (transposed tile)

    const int tid  = threadIdx.x;
    const int lane = tid & 31;
    const int wave = tid >> 5;
    const int kh   = lane >> 4;       // K-half selector within fragment
    const int ln   = lane & 15;

    const int n0 = blockIdx.x * 128;
    const int m0 = blockIdx.y * 128;
    const int wm = (wave & 3) * 32;
    const int wn = (wave >> 2) * 64;

    // Per-thread staging assignment: 2 x 16B chunks of A, 2 x 16B chunks of B.
    const int aRow  = tid >> 2;            // rows aRow and aRow+64
    const int aKoff = (tid & 3) * 8;
    const int bK    = tid >> 4;            // k-lines bK and bK+16
    const int bNoff = (tid & 15) * 8;

    const u16* aSrc0 = A + (size_t)(m0 + aRow) * HID + aKoff;
    const u16* aSrc1 = aSrc0 + (size_t)64 * HID;
    const u16* bSrc0 = W + (size_t)bK * HID + n0 + bNoff;
    const u16* bSrc1 = bSrc0 + (size_t)16 * HID;

    const unsigned ldsA0 = (unsigned)(size_t)&As[0][aRow][aKoff];
    const unsigned ldsA1 = (unsigned)(size_t)&As[0][aRow + 64][aKoff];
    const unsigned bufStride = (unsigned)sizeof(As[0]);

    uint4 rb0, rb1;

    auto issueA = [&](int buf, int kt) {
        async_g2l_b128(ldsA0 + buf * bufStride, aSrc0 + kt * 32);
        async_g2l_b128(ldsA1 + buf * bufStride, aSrc1 + kt * 32);
    };
    auto loadB = [&](int kt) {
        rb0 = *reinterpret_cast<const uint4*>(bSrc0 + (size_t)kt * 32 * HID);
        rb1 = *reinterpret_cast<const uint4*>(bSrc1 + (size_t)kt * 32 * HID);
    };
    auto storeB = [&](int buf) {
        union { u16 e[8]; uint4 v; } t0, t1;
        t0.v = rb0; t1.v = rb1;
        for (int j = 0; j < 8; ++j) {
            Bs[buf][bNoff + j][bK]      = t0.e[j];
            Bs[buf][bNoff + j][bK + 16] = t1.e[j];
        }
    };

    v8f acc[2][4];
    const v8f zf = {0.f, 0.f, 0.f, 0.f, 0.f, 0.f, 0.f, 0.f};
    for (int mi = 0; mi < 2; ++mi)
        for (int ni = 0; ni < 4; ++ni)
            acc[mi][ni] = zf;

    // Prologue: stage tile 0.
    issueA(0, 0);
    loadB(0);
    storeB(0);
    wait_asynccnt0();
    __syncthreads();

    const int NT = HID / 32;
    for (int kt = 0; kt < NT; ++kt) {
        const int cur  = kt & 1;
        const int nxt  = cur ^ 1;
        const bool more = (kt + 1) < NT;

        // Issue next tile's data movement first (overlaps WMMA below).
        if (more) { issueA(nxt, kt + 1); loadB(kt + 1); }

        // ---- fragments + 8 WMMAs per wave on current buffer ----
        v16bf af[2];
        for (int mi = 0; mi < 2; ++mi) {
            int row = wm + mi * 16 + ln;
            af[mi] = frag_from2(
                *reinterpret_cast<const uint4*>(&As[cur][row][kh * 8]),
                *reinterpret_cast<const uint4*>(&As[cur][row][kh * 8 + 16]));
        }
        for (int ni = 0; ni < 4; ++ni) {
            int n = wn + ni * 16 + ln;
            v16bf bf = frag_from2(
                *reinterpret_cast<const uint4*>(&Bs[cur][n][kh * 8]),
                *reinterpret_cast<const uint4*>(&Bs[cur][n][kh * 8 + 16]));
            for (int mi = 0; mi < 2; ++mi)
                acc[mi][ni] = wmma_bf16(af[mi], bf, acc[mi][ni]);
        }

        if (more) storeB(nxt);
        wait_asynccnt0();
        __syncthreads();
    }

    // ---- epilogue: C fragment row = r + 8*(lane/16), col = lane%16 ----
    for (int mi = 0; mi < 2; ++mi)
        for (int ni = 0; ni < 4; ++ni)
            for (int r = 0; r < 8; ++r) {
                int row = m0 + wm + mi * 16 + r + 8 * kh;
                int col = n0 + wn + ni * 16 + ln;
                float v = acc[mi][ni][r];
                if (out_f32) Cf[(size_t)row * HID + col] = v;
                else         Cb[(size_t)row * HID + col] = f2bf(v);
            }
}

// ---------------------------------------------------------------------------
// Kernel 3: RoPE in-place on a bf16 [NTOK][HID] buffer (token-major).
// ---------------------------------------------------------------------------
__global__ __launch_bounds__(256) void rope_kernel(u16* __restrict__ buf) {
    int idx = blockIdx.x * blockDim.x + threadIdx.x;   // NTOK*NHEAD*64 threads
    if (idx >= NTOK * NHEAD * 64) return;
    int j   = idx & 63;
    int h   = (idx >> 6) & (NHEAD - 1);
    int tok = idx >> 10;
    int pos = tok & (SEQ - 1);

    float inv = __powf(10000.f, -(float)j * (1.0f / 64.0f));
    float ang = (float)pos * inv;
    float s, c;
    __sincosf(ang, &s, &c);

    size_t base = (size_t)tok * HID + h * HDIM + j;
    float x1 = bf2f(buf[base]);
    float x2 = bf2f(buf[base + 64]);
    buf[base]      = f2bf(x1 * c - x2 * s);
    buf[base + 64] = f2bf(x2 * c + x1 * s);
}

// ---------------------------------------------------------------------------
// Kernel 4: causal flash attention.
//   grid = (S/64, NHEAD, B), block = 128 (4 waves).  Each wave: 16 q rows.
//   Q fragments persistent in VGPRs; K tile and transposed V tile in LDS;
//   P re-layout (C-frag -> A-frag) via per-wave LDS scratch.
// ---------------------------------------------------------------------------
__global__ __launch_bounds__(128) void flash_attn(const u16* __restrict__ Qb,
                                                  const u16* __restrict__ Kb,
                                                  const u16* __restrict__ Vb,
                                                  u16* __restrict__ Ob) {
    __shared__ alignas(16) u16 Kt[64][136];      // [key][d]
    __shared__ alignas(16) u16 Vt[128][72];      // [d][key] (transposed)
    __shared__ alignas(16) u16 Ps[4][16][72];    // per-wave P scratch [row][key]

    const int tid  = threadIdx.x;
    const int lane = tid & 31;
    const int wave = tid >> 5;
    const int kh   = lane >> 4;
    const int ln   = lane & 15;

    const int q0   = blockIdx.x * 64;
    const int head = blockIdx.y;
    const int b    = blockIdx.z;
    const float scale = 0.08838834764831845f;    // 1/sqrt(128)

    // ---- persistent Q fragments: 16 rows x 128 d per wave (4 x 16x32) ----
    const u16* qptr = Qb + ((size_t)(b * SEQ + q0 + wave * 16 + ln)) * HID + head * HDIM;
    v16bf qf[4];
    for (int g = 0; g < 4; ++g)
        qf[g] = frag_from2(*reinterpret_cast<const uint4*>(&qptr[g * 32 + kh * 8]),
                           *reinterpret_cast<const uint4*>(&qptr[g * 32 + kh * 8 + 16]));

    float mst[8], lst[8];
    v8f oacc[8];
    const v8f zf = {0.f, 0.f, 0.f, 0.f, 0.f, 0.f, 0.f, 0.f};
    for (int r = 0; r < 8; ++r) { mst[r] = -3.0e38f; lst[r] = 0.f; }
    for (int d = 0; d < 8; ++d) oacc[d] = zf;

    const int nkt = q0 / 64 + 1;                 // causal tile bound
    for (int kt = 0; kt < nkt; ++kt) {
        __syncthreads();                          // protect previous tile reads
        // ---- stage K tile (row-major) and V tile (transposed) ----
        {
            int key = tid >> 1;
            int dh  = (tid & 1) * 64;
            const u16* ksrc = Kb + ((size_t)(b * SEQ + kt * 64 + key)) * HID + head * HDIM + dh;
            const u16* vsrc = Vb + ((size_t)(b * SEQ + kt * 64 + key)) * HID + head * HDIM + dh;
            for (int j = 0; j < 8; ++j) {
                *reinterpret_cast<uint4*>(&Kt[key][dh + j * 8]) =
                    *reinterpret_cast<const uint4*>(&ksrc[j * 8]);
                union { u16 e[8]; uint4 v; } t;
                t.v = *reinterpret_cast<const uint4*>(&vsrc[j * 8]);
                for (int e = 0; e < 8; ++e) Vt[dh + j * 8 + e][key] = t.e[e];
            }
        }
        __syncthreads();

        // ---- S = Q K^T : 16x64 per wave (4 accumulators, K=128) ----
        v8f sa[4];
        for (int ni = 0; ni < 4; ++ni) {
            v8f s = zf;
            int n = ni * 16 + ln;                 // key column
            for (int g = 0; g < 4; ++g) {
                v16bf bfrag = frag_from2(
                    *reinterpret_cast<const uint4*>(&Kt[n][g * 32 + kh * 8]),
                    *reinterpret_cast<const uint4*>(&Kt[n][g * 32 + kh * 8 + 16]));
                s = wmma_bf16(qf[g], bfrag, s);
            }
            sa[ni] = s;
        }

        // ---- scale + causal mask (only on diagonal tile) ----
        const bool need_mask = (kt * 64 == q0);
        for (int ni = 0; ni < 4; ++ni)
            for (int r = 0; r < 8; ++r) {
                float v = sa[ni][r] * scale;
                if (need_mask) {
                    int qrow = q0 + wave * 16 + r + 8 * kh;
                    int kcol = kt * 64 + ni * 16 + ln;
                    if (kcol > qrow) v = -1.0e30f;
                }
                sa[ni][r] = v;
            }

        // ---- online softmax: row stats across 16 lanes (half-wave) ----
        for (int r = 0; r < 8; ++r) {
            float mrow = sa[0][r];
            mrow = fmaxf(mrow, sa[1][r]);
            mrow = fmaxf(mrow, sa[2][r]);
            mrow = fmaxf(mrow, sa[3][r]);
            for (int off = 1; off < 16; off <<= 1)
                mrow = fmaxf(mrow, __shfl_xor(mrow, off, 32));
            float mnew  = fmaxf(mst[r], mrow);
            float alpha = __expf(mst[r] - mnew);
            float rsum  = 0.f;
            for (int ni = 0; ni < 4; ++ni) {
                float p = __expf(sa[ni][r] - mnew);
                sa[ni][r] = p;
                rsum += p;
            }
            for (int off = 1; off < 16; off <<= 1)
                rsum += __shfl_xor(rsum, off, 32);
            lst[r] = lst[r] * alpha + rsum;
            mst[r] = mnew;
            for (int d = 0; d < 8; ++d) oacc[d][r] *= alpha;
        }

        // ---- re-layout P: C-fragment -> LDS -> A-fragment (wave-local) ----
        for (int ni = 0; ni < 4; ++ni)
            for (int r = 0; r < 8; ++r)
                Ps[wave][r + 8 * kh][ni * 16 + ln] = f2bf(sa[ni][r]);
        v16bf pf[2];
        for (int g = 0; g < 2; ++g)
            pf[g] = frag_from2(
                *reinterpret_cast<const uint4*>(&Ps[wave][ln][g * 32 + kh * 8]),
                *reinterpret_cast<const uint4*>(&Ps[wave][ln][g * 32 + kh * 8 + 16]));

        // ---- O += P V : 16x128 per wave (8 accumulators, K=64) ----
        for (int d = 0; d < 8; ++d) {
            v8f o = oacc[d];
            int n = d * 16 + ln;                  // output-d column
            for (int g = 0; g < 2; ++g) {
                v16bf bfrag = frag_from2(
                    *reinterpret_cast<const uint4*>(&Vt[n][g * 32 + kh * 8]),
                    *reinterpret_cast<const uint4*>(&Vt[n][g * 32 + kh * 8 + 16]));
                o = wmma_bf16(pf[g], bfrag, o);
            }
            oacc[d] = o;
        }
    }

    // ---- normalize and store bf16 attention output (token-major) ----
    for (int r = 0; r < 8; ++r) {
        float rinv = 1.0f / lst[r];
        size_t tok = (size_t)(b * SEQ + q0 + wave * 16 + r + 8 * kh);
        for (int d = 0; d < 8; ++d)
            Ob[tok * HID + head * HDIM + d * 16 + ln] = f2bf(oacc[d][r] * rinv);
    }
}

// ---------------------------------------------------------------------------
// Host-side launcher
// ---------------------------------------------------------------------------
extern "C" void kernel_launch(void* const* d_in, const int* in_sizes, int n_in,
                              void* d_out, int out_size, void* d_ws, size_t ws_size,
                              hipStream_t stream) {
    (void)in_sizes; (void)n_in; (void)out_size; (void)ws_size;

    const float* hs = (const float*)d_in[0];
    const float* Wq = (const float*)d_in[1];
    const float* Wk = (const float*)d_in[2];
    const float* Wv = (const float*)d_in[3];
    const float* Wo = (const float*)d_in[4];
    // d_in[5] = attention_mask: causal, handled analytically.
    float* out = (float*)d_out;

    char* ws = (char*)d_ws;
    const size_t ACT = (size_t)NTOK * HID * 2;   // 16 MB bf16 activations
    const size_t WGT = (size_t)HID * HID * 2;    // 8 MB bf16 weights
    u16* Xb  = (u16*)(ws);
    u16* Wqb = (u16*)(ws + ACT);
    u16* Wkb = (u16*)(ws + ACT + WGT);
    u16* Wvb = (u16*)(ws + ACT + 2 * WGT);
    u16* Wob = (u16*)(ws + ACT + 3 * WGT);
    u16* Qb  = (u16*)(ws + ACT + 4 * WGT);
    u16* Kb  = (u16*)(ws + 2 * ACT + 4 * WGT);
    u16* Vb  = (u16*)(ws + 3 * ACT + 4 * WGT);
    u16* Ab  = (u16*)(ws + 4 * ACT + 4 * WGT);

    // 1) fp32 -> bf16
    {
        int n8a = (NTOK * HID) / 8, n8w = (HID * HID) / 8;
        cvt_f32_bf16<<<(n8a + 255) / 256, 256, 0, stream>>>(hs, Xb, n8a);
        cvt_f32_bf16<<<(n8w + 255) / 256, 256, 0, stream>>>(Wq, Wqb, n8w);
        cvt_f32_bf16<<<(n8w + 255) / 256, 256, 0, stream>>>(Wk, Wkb, n8w);
        cvt_f32_bf16<<<(n8w + 255) / 256, 256, 0, stream>>>(Wv, Wvb, n8w);
        cvt_f32_bf16<<<(n8w + 255) / 256, 256, 0, stream>>>(Wo, Wob, n8w);
    }

    // 2) Q/K/V projections (WMMA GEMM, bf16 out)
    dim3 ggrid(HID / 128, NTOK / 128);
    gemm_bf16<<<ggrid, 256, 0, stream>>>(Xb, Wqb, Qb, nullptr, 0);
    gemm_bf16<<<ggrid, 256, 0, stream>>>(Xb, Wkb, Kb, nullptr, 0);
    gemm_bf16<<<ggrid, 256, 0, stream>>>(Xb, Wvb, Vb, nullptr, 0);

    // 3) RoPE on Q and K
    {
        int nthr = NTOK * NHEAD * 64;
        rope_kernel<<<(nthr + 255) / 256, 256, 0, stream>>>(Qb);
        rope_kernel<<<(nthr + 255) / 256, 256, 0, stream>>>(Kb);
    }

    // 4) causal flash attention (WMMA)
    flash_attn<<<dim3(SEQ / 64, NHEAD, 2), 128, 0, stream>>>(Qb, Kb, Vb, Ab);

    // 5) output projection (WMMA GEMM, f32 out)
    gemm_bf16<<<ggrid, 256, 0, stream>>>(Ab, Wob, nullptr, out, 1);
}
